// DecoderCell_59742995087471
// MI455X (gfx1250) — matrix operand order repaired and checked by reading, not compiled, not run on hardware
//
#include <hip/hip_runtime.h>
#include <stdint.h>

typedef _Float16 h16;
typedef __attribute__((ext_vector_type(16))) _Float16 v16h;
typedef __attribute__((ext_vector_type(8)))  _Float16 v8h;
typedef __attribute__((ext_vector_type(8)))  float    v8f;
typedef uint32_t u32x4 __attribute__((ext_vector_type(4)));
typedef uint32_t u32x8 __attribute__((ext_vector_type(8)));

#define D_MODEL 1024
#define NBATCH  2
#define SEQLEN  2048
#define NHEADS  16
#define HEADD   64
#define MROWS   (NBATCH * SEQLEN)

__device__ __forceinline__ v16h cat8(v8h lo, v8h hi) {
  return __builtin_shufflevector(lo, hi, 0, 1, 2, 3, 4, 5, 6, 7,
                                         8, 9, 10, 11, 12, 13, 14, 15);
}

__device__ __forceinline__ uint32_t lds_off(const void* p) {
  // LDS aperture flat address: low 32 bits are the LDS byte offset
  return (uint32_t)(uintptr_t)p;
}

// ---------------------------------------------------------------------------
// TDM: DMA one 2-D f16 tile (tile_w x tile_h elements) from global into LDS.
// D# per CDNA5 ISA ch.8: group0 = {count/flags, lds_addr, global_addr, type},
// group1 = {mask/size/pad, dims, tile dims, dim0 stride}. pad: +16B per 64B
// row so LDS rows are 80B (= 40 h16) apart, matching the padded tiles below.
// ---------------------------------------------------------------------------
__device__ __forceinline__ void tdm_load_tile_f16(
    uint32_t lds_addr, const h16* gtile,
    uint32_t tensor_w, uint32_t tensor_h, uint32_t row_stride_elems,
    uint32_t tile_w, uint32_t tile_h) {
  uint64_t ga = (uint64_t)(uintptr_t)gtile;
  u32x4 g0;
  g0[0] = 1u;                                                  // count=1 (valid)
  g0[1] = lds_addr;                                            // lds_addr[31:0]
  g0[2] = (uint32_t)ga;                                        // global_addr lo
  g0[3] = ((uint32_t)(ga >> 32) & 0x01FFFFFFu) | (2u << 30);   // addr hi | type=2
  u32x8 g1;
  g1[0] = (1u << 16)          // data_size = 1 -> 2 bytes
        | (1u << 20)          // pad_enable
        | (3u << 22)          // pad_interval code 3 = 16 DWORDs (64 B)
        | (3u << 25);         // pad_amount  code 3 =  4 DWORDs (16 B)
  g1[1] = (tensor_w & 0xFFFFu) << 16;                          // tensor_dim0 lo
  g1[2] = (tensor_w >> 16) | ((tensor_h & 0xFFFFu) << 16);     // dim0 hi | dim1 lo
  g1[3] = (tensor_h >> 16) | ((tile_w & 0xFFFFu) << 16);       // dim1 hi | tile_dim0
  g1[4] = tile_h & 0xFFFFu;                                    // tile_dim1, tile_dim2=0
  g1[5] = row_stride_elems;                                    // tensor_dim0_stride lo
  g1[6] = 0u;                                                  // stride hi, dim1_stride=0
  g1[7] = 0u;
  asm volatile("tensor_load_to_lds %0, %1" :: "s"(g0), "s"(g1) : "memory");
}

__device__ __forceinline__ void async_ld_b128(uint32_t lds_addr,
                                              const void* base, uint32_t voff) {
  asm volatile("global_load_async_to_lds_b128 %0, %1, %2 offset:0"
               :: "v"(lds_addr), "v"(voff), "s"((uint64_t)(uintptr_t)base)
               : "memory");
}
__device__ __forceinline__ void wait_async0() {
  asm volatile("s_wait_asynccnt 0x0" ::: "memory");
}

// ---------------------------------------------------------------------------
// Elementwise f32 -> f16 convert (4 elements / thread).
// ---------------------------------------------------------------------------
__global__ __launch_bounds__(256)
void cvt_f16_kernel(const float* __restrict__ X, h16* __restrict__ Y, int n) {
  int i = (blockIdx.x * 256 + threadIdx.x) * 4;
  if (i + 3 < n) {
#pragma unroll
    for (int j = 0; j < 4; ++j) Y[i + j] = (h16)X[i + j];
  }
}

// ---------------------------------------------------------------------------
// Wt[n*K + k] = (f16) W[k*N + n]  — 32x32 tiles through LDS.
// ---------------------------------------------------------------------------
__global__ __launch_bounds__(256)
void tr_cvt_kernel(const float* __restrict__ W, h16* __restrict__ Wt,
                   int K, int N) {
  __shared__ float tile[32][33];
  const int kb = blockIdx.y * 32, nb = blockIdx.x * 32;
  const int tx = threadIdx.x & 31, ty = threadIdx.x >> 5;   // 32 x 8
#pragma unroll
  for (int i = 0; i < 32; i += 8)
    tile[ty + i][tx] = W[(size_t)(kb + ty + i) * N + (nb + tx)];
  __syncthreads();
#pragma unroll
  for (int i = 0; i < 32; i += 8)
    Wt[(size_t)(nb + ty + i) * K + (kb + tx)] = (h16)tile[tx][ty + i];
}

// ---------------------------------------------------------------------------
// GEMM: C[M,N] (f16) = A[M,K] (f16) @ Wt[N,K]^T (f16) + bias (f32).
// 128 threads / 4 waves, block tile 64x64, wave tile 32x32 (2x2 WMMA).
// Tiles staged by TDM (double buffered, s_wait_tensorcnt).
// ---------------------------------------------------------------------------
__global__ __launch_bounds__(128)
void gemm_f16_kernel(const h16* __restrict__ A, const h16* __restrict__ Wt,
                     const float* __restrict__ bias, h16* __restrict__ C,
                     int M, int N, int K) {
  __shared__ h16 As[2][64][40];   // padded 80B rows (TDM pad feature)
  __shared__ h16 Bt[2][64][40];

  const int tid  = threadIdx.x;
  const int lane = tid & 31;
  const int lr   = lane & 15;
  const int lh   = lane >> 4;
  const int moff = ((tid >> 5) >> 1) * 32;
  const int noff = ((tid >> 5) & 1) * 32;
  const int mbase = blockIdx.y * 64;
  const int nbase = blockIdx.x * 64;
  const int nkt = K / 32;

  const uint32_t aLds[2] = { lds_off(&As[0][0][0]), lds_off(&As[1][0][0]) };
  const uint32_t bLds[2] = { lds_off(&Bt[0][0][0]), lds_off(&Bt[1][0][0]) };
  const bool issuer = (tid < 32);          // wave 0 issues the DMAs

  if (issuer) {
    tdm_load_tile_f16(aLds[0], A  + (size_t)mbase * K, K, M, K, 32, 64);
    tdm_load_tile_f16(bLds[0], Wt + (size_t)nbase * K, K, N, K, 32, 64);
  }

  v8f acc[2][2] = {};
  for (int kt = 0; kt < nkt; ++kt) {
    const int cur = kt & 1;
    if (issuer) {
      if (kt + 1 < nkt) {
        const int nxt = cur ^ 1;
        tdm_load_tile_f16(aLds[nxt], A  + (size_t)mbase * K + (kt + 1) * 32,
                          K, M, K, 32, 64);
        tdm_load_tile_f16(bLds[nxt], Wt + (size_t)nbase * K + (kt + 1) * 32,
                          K, N, K, 32, 64);
        __builtin_amdgcn_s_wait_tensorcnt(2);   // current tile complete
      } else {
        __builtin_amdgcn_s_wait_tensorcnt(0);
      }
    }
    __syncthreads();

    v16h af[2], bf[2];
#pragma unroll
    for (int mi = 0; mi < 2; ++mi) {
      const h16* p = &As[cur][moff + mi * 16 + lr][lh * 8];
      af[mi] = cat8(*(const v8h*)p, *(const v8h*)(p + 16));
    }
#pragma unroll
    for (int ni = 0; ni < 2; ++ni) {
      const h16* p = &Bt[cur][noff + ni * 16 + lr][lh * 8];
      bf[ni] = cat8(*(const v8h*)p, *(const v8h*)(p + 16));
    }
#pragma unroll
    for (int mi = 0; mi < 2; ++mi)
#pragma unroll
      for (int ni = 0; ni < 2; ++ni)
        acc[mi][ni] = __builtin_amdgcn_wmma_f32_16x16x32_f16(
            false, af[mi], false, bf[ni], (short)0, acc[mi][ni], false, false);
    __syncthreads();
  }

#pragma unroll
  for (int mi = 0; mi < 2; ++mi)
#pragma unroll
    for (int ni = 0; ni < 2; ++ni)
#pragma unroll
      for (int r = 0; r < 8; ++r) {
        int gm = mbase + moff + mi * 16 + lh * 8 + r;
        int gn = nbase + noff + ni * 16 + lr;
        C[(size_t)gm * N + gn] = (h16)(acc[mi][ni][r] + bias[gn]);
      }
}

// ---------------------------------------------------------------------------
// Flash attention (f16 in/out): block = (batch, head, 64-query tile).
// Q/K tiles arrive via global_load_async_to_lds_b128; V is transposed on the
// fly (b128 loads + b16 DS scatter) to serve as B^T of the PV matmul.
// ---------------------------------------------------------------------------
__global__ __launch_bounds__(128)
void attn_f16_kernel(const h16* __restrict__ Q, const h16* __restrict__ Kp,
                     const h16* __restrict__ V, h16* __restrict__ O,
                     int Lk, int causal) {
  __shared__ h16 Qs[64][72];    // [q][d]
  __shared__ h16 Ks[64][72];    // [k][d]   (== B^T for Q*K^T)
  __shared__ h16 Vt[64][72];    // [d][k]   (== B^T for P*V)
  __shared__ h16 Ps[64][72];    // [q][k] probabilities
  __shared__ float Ss[64][65];  // raw scores
  __shared__ float rowm[64], rowl[64], rowsc[64];

  const int tid  = threadIdx.x;
  const int lane = tid & 31;
  const int lr   = lane & 15;
  const int lh   = lane >> 4;
  const int moff = ((tid >> 5) >> 1) * 32;
  const int noff = ((tid >> 5) & 1) * 32;
  const int qt = blockIdx.x, hh = blockIdx.y, bb = blockIdx.z;
  const int qb = qt * 64;
  const size_t qoff   = ((size_t)bb * SEQLEN + qb) * D_MODEL + hh * HEADD;
  const size_t kvbase = (size_t)bb * Lk * D_MODEL + hh * HEADD;

  const uint32_t qsBase = lds_off(&Qs[0][0]);
  const uint32_t ksBase = lds_off(&Ks[0][0]);

  // Q tile: 64 rows x 128 B = 512 x 16-B chunks, 4 per thread, async
#pragma unroll
  for (int i = 0; i < 4; ++i) {
    int idx = i * 128 + tid;
    int r = idx >> 3, c = idx & 7;
    async_ld_b128(qsBase + r * 144 + c * 16, Q + qoff,
                  (uint32_t)(r * (D_MODEL * 2) + c * 16));
  }
  if (tid < 64) { rowm[tid] = -3.0e38f; rowl[tid] = 0.0f; }

  v8f oacc[2][2] = {};
  const int nkt = causal ? (qt + 1) : (Lk / 64);

  for (int kt = 0; kt < nkt; ++kt) {
    const int kb = kt * 64;
    // K tile via async DMA
#pragma unroll
    for (int i = 0; i < 4; ++i) {
      int idx = i * 128 + tid;
      int r = idx >> 3, c = idx & 7;
      async_ld_b128(ksBase + r * 144 + c * 16, Kp + kvbase,
                    (uint32_t)((kb + r) * (D_MODEL * 2) + c * 16));
    }
    // V tile: load 8 f16 and scatter transposed
#pragma unroll
    for (int i = 0; i < 4; ++i) {
      int idx = i * 128 + tid;
      int r = idx >> 3, c = idx & 7;
      v8h vv = *(const v8h*)(V + kvbase + (size_t)(kb + r) * D_MODEL + c * 8);
#pragma unroll
      for (int j = 0; j < 8; ++j) Vt[c * 8 + j][r] = vv[j];
    }
    wait_async0();
    __syncthreads();

    // S = (Q K^T) / sqrt(64)
    v8f sacc[2][2] = {};
#pragma unroll
    for (int kd = 0; kd < 64; kd += 32) {
      v16h af[2], bf[2];
#pragma unroll
      for (int mi = 0; mi < 2; ++mi) {
        const h16* p = &Qs[moff + mi * 16 + lr][kd + lh * 8];
        af[mi] = cat8(*(const v8h*)p, *(const v8h*)(p + 16));
      }
#pragma unroll
      for (int ni = 0; ni < 2; ++ni) {
        const h16* p = &Ks[noff + ni * 16 + lr][kd + lh * 8];
        bf[ni] = cat8(*(const v8h*)p, *(const v8h*)(p + 16));
      }
#pragma unroll
      for (int mi = 0; mi < 2; ++mi)
#pragma unroll
        for (int ni = 0; ni < 2; ++ni)
          sacc[mi][ni] = __builtin_amdgcn_wmma_f32_16x16x32_f16(
              false, af[mi], false, bf[ni], (short)0, sacc[mi][ni], false, false);
    }
#pragma unroll
    for (int mi = 0; mi < 2; ++mi)
#pragma unroll
      for (int ni = 0; ni < 2; ++ni)
#pragma unroll
        for (int r = 0; r < 8; ++r) {
          int qrow = moff + mi * 16 + lh * 8 + r;
          int kcol = noff + ni * 16 + lr;
          float s = sacc[mi][ni][r] * 0.125f;
          if (causal && (kb + kcol) > (qb + qrow)) s = -1.0e9f;
          Ss[qrow][kcol] = s;
        }
    __syncthreads();

    // online softmax, one thread per query row
    if (tid < 64) {
      float mold = rowm[tid];
      float mx = mold;
#pragma unroll 8
      for (int j = 0; j < 64; ++j) mx = fmaxf(mx, Ss[tid][j]);
      float sc = __expf(mold - mx);
      float l = rowl[tid] * sc;
#pragma unroll 8
      for (int j = 0; j < 64; ++j) {
        float p = __expf(Ss[tid][j] - mx);
        l += p;
        Ps[tid][j] = (h16)p;
      }
      rowm[tid] = mx; rowl[tid] = l; rowsc[tid] = sc;
    }
    __syncthreads();

    // rescale O, then O += P @ V
#pragma unroll
    for (int mi = 0; mi < 2; ++mi)
#pragma unroll
      for (int r = 0; r < 8; ++r) {
        float sc = rowsc[moff + mi * 16 + lh * 8 + r];
        oacc[mi][0][r] *= sc;
        oacc[mi][1][r] *= sc;
      }
#pragma unroll
    for (int kk = 0; kk < 64; kk += 32) {
      v16h af[2], bf[2];
#pragma unroll
      for (int mi = 0; mi < 2; ++mi) {
        const h16* p = &Ps[moff + mi * 16 + lr][kk + lh * 8];
        af[mi] = cat8(*(const v8h*)p, *(const v8h*)(p + 16));
      }
#pragma unroll
      for (int ni = 0; ni < 2; ++ni) {
        const h16* p = &Vt[noff + ni * 16 + lr][kk + lh * 8];
        bf[ni] = cat8(*(const v8h*)p, *(const v8h*)(p + 16));
      }
#pragma unroll
      for (int mi = 0; mi < 2; ++mi)
#pragma unroll
        for (int ni = 0; ni < 2; ++ni)
          oacc[mi][ni] = __builtin_amdgcn_wmma_f32_16x16x32_f16(
              false, af[mi], false, bf[ni], (short)0, oacc[mi][ni], false, false);
    }
    __syncthreads();
  }

#pragma unroll
  for (int mi = 0; mi < 2; ++mi)
#pragma unroll
    for (int ni = 0; ni < 2; ++ni)
#pragma unroll
      for (int r = 0; r < 8; ++r) {
        int qrow = moff + mi * 16 + lh * 8 + r;
        int dcol = noff + ni * 16 + lr;
        O[qoff + (size_t)qrow * D_MODEL + dcol] = (h16)(oacc[mi][ni][r] / rowl[qrow]);
      }
}

// ---------------------------------------------------------------------------
// out = LayerNorm(X + Y); X f32 residual, Y f16; writes f32 and f16 copies.
// ---------------------------------------------------------------------------
__global__ __launch_bounds__(256)
void add_ln_kernel(const float* __restrict__ X, const h16* __restrict__ Y,
                   const float* __restrict__ g, const float* __restrict__ be,
                   float* __restrict__ outf, h16* __restrict__ outh) {
  __shared__ float red[256];
  __shared__ float red2[256];
  __shared__ float s_mean, s_rstd;
  const int row = blockIdx.x, tid = threadIdx.x;
  const size_t base = (size_t)row * D_MODEL;

  float v[4];
  float s = 0.f, s2 = 0.f;
#pragma unroll
  for (int i = 0; i < 4; ++i) {
    int c = tid + i * 256;
    float x = X[base + c] + (float)Y[base + c];
    v[i] = x; s += x; s2 += x * x;
  }
  red[tid] = s; red2[tid] = s2;
  __syncthreads();
  for (int off = 128; off > 0; off >>= 1) {
    if (tid < off) { red[tid] += red[tid + off]; red2[tid] += red2[tid + off]; }
    __syncthreads();
  }
  if (tid == 0) {
    float mean = red[0] * (1.0f / D_MODEL);
    float var  = red2[0] * (1.0f / D_MODEL) - mean * mean;
    s_mean = mean;
    s_rstd = rsqrtf(var + 1e-5f);
  }
  __syncthreads();
#pragma unroll
  for (int i = 0; i < 4; ++i) {
    int c = tid + i * 256;
    float o = g[c] * (v[i] - s_mean) * s_rstd + be[c];
    outf[base + c] = o;
    outh[base + c] = (h16)o;
  }
}

// ---------------------------------------------------------------------------
extern "C" void kernel_launch(void* const* d_in, const int* in_sizes, int n_in,
                              void* d_out, int out_size, void* d_ws, size_t ws_size,
                              hipStream_t stream) {
  (void)in_sizes; (void)n_in; (void)out_size; (void)ws_size;
  const float* S0  = (const float*)d_in[0];
  const float* Hf  = (const float*)d_in[1];
  const float* Wq1 = (const float*)d_in[2];  const float* bq1 = (const float*)d_in[3];
  const float* Wk1 = (const float*)d_in[4];  const float* bk1 = (const float*)d_in[5];
  const float* Wv1 = (const float*)d_in[6];  const float* bv1 = (const float*)d_in[7];
  const float* Wo1 = (const float*)d_in[8];  const float* bo1 = (const float*)d_in[9];
  const float* g1  = (const float*)d_in[10]; const float* be1 = (const float*)d_in[11];
  const float* Wq2 = (const float*)d_in[12]; const float* bq2 = (const float*)d_in[13];
  const float* Wk2 = (const float*)d_in[14]; const float* bk2 = (const float*)d_in[15];
  const float* Wv2 = (const float*)d_in[16]; const float* bv2 = (const float*)d_in[17];
  const float* Wo2 = (const float*)d_in[18]; const float* bo2 = (const float*)d_in[19];
  const float* g2  = (const float*)d_in[20]; const float* be2 = (const float*)d_in[21];
  const float* Wf  = (const float*)d_in[22]; const float* bfb = (const float*)d_in[23];
  const float* g3  = (const float*)d_in[24]; const float* be3 = (const float*)d_in[25];

  const size_t ACT = (size_t)MROWS * D_MODEL;          // activation elements
  const size_t WEL = (size_t)D_MODEL * D_MODEL;        // weight elements
  uint8_t* w = (uint8_t*)d_ws;
  h16* S0h = (h16*)w;            w += ACT * 2;
  h16* Hh  = (h16*)w;            w += ACT * 2;
  h16* Wt[9];
  for (int i = 0; i < 9; ++i) { Wt[i] = (h16*)w; w += WEL * 2; }
  h16* Qh  = (h16*)w;            w += ACT * 2;
  h16* Kh  = (h16*)w;            w += ACT * 2;
  h16* Vh  = (h16*)w;            w += ACT * 2;
  h16* Cxh = (h16*)w;            w += ACT * 2;
  h16* Pjh = (h16*)w;            w += ACT * 2;
  h16* S1h = (h16*)w;            w += ACT * 2;
  h16* S2h = (h16*)w;            w += ACT * 2;
  float* S1f = (float*)w;        w += ACT * 4;
  float* S2f = (float*)w;        w += ACT * 4;
  float* S3f = (float*)d_out;

  dim3 cblk(256), cgrid((unsigned)(ACT / 1024));
  dim3 tblk(256), tgrid(D_MODEL / 32, D_MODEL / 32);
  dim3 gblk(128), ggrid(D_MODEL / 64, MROWS / 64);
  dim3 ablk(128), agrid(SEQLEN / 64, NHEADS, NBATCH);
  dim3 lblk(256), lgrid(MROWS);

  // one-time precision/layout conversion
  cvt_f16_kernel<<<cgrid, cblk, 0, stream>>>(S0, S0h, (int)ACT);
  cvt_f16_kernel<<<cgrid, cblk, 0, stream>>>(Hf, Hh, (int)ACT);
  const float* Wsrc[9] = {Wq1, Wk1, Wv1, Wo1, Wq2, Wk2, Wv2, Wo2, Wf};
  for (int i = 0; i < 9; ++i)
    tr_cvt_kernel<<<tgrid, tblk, 0, stream>>>(Wsrc[i], Wt[i], D_MODEL, D_MODEL);

  // ---- masked self-attention + add&norm ----
  gemm_f16_kernel<<<ggrid, gblk, 0, stream>>>(S0h, Wt[0], bq1, Qh, MROWS, D_MODEL, D_MODEL);
  gemm_f16_kernel<<<ggrid, gblk, 0, stream>>>(S0h, Wt[1], bk1, Kh, MROWS, D_MODEL, D_MODEL);
  gemm_f16_kernel<<<ggrid, gblk, 0, stream>>>(S0h, Wt[2], bv1, Vh, MROWS, D_MODEL, D_MODEL);
  attn_f16_kernel<<<agrid, ablk, 0, stream>>>(Qh, Kh, Vh, Cxh, SEQLEN, 1);
  gemm_f16_kernel<<<ggrid, gblk, 0, stream>>>(Cxh, Wt[3], bo1, Pjh, MROWS, D_MODEL, D_MODEL);
  add_ln_kernel<<<lgrid, lblk, 0, stream>>>(S0, Pjh, g1, be1, S1f, S1h);

  // ---- cross-attention over encoder output + add&norm ----
  gemm_f16_kernel<<<ggrid, gblk, 0, stream>>>(S1h, Wt[4], bq2, Qh, MROWS, D_MODEL, D_MODEL);
  gemm_f16_kernel<<<ggrid, gblk, 0, stream>>>(Hh,  Wt[5], bk2, Kh, MROWS, D_MODEL, D_MODEL);
  gemm_f16_kernel<<<ggrid, gblk, 0, stream>>>(Hh,  Wt[6], bv2, Vh, MROWS, D_MODEL, D_MODEL);
  attn_f16_kernel<<<agrid, ablk, 0, stream>>>(Qh, Kh, Vh, Cxh, SEQLEN, 0);
  gemm_f16_kernel<<<ggrid, gblk, 0, stream>>>(Cxh, Wt[7], bo2, Pjh, MROWS, D_MODEL, D_MODEL);
  add_ln_kernel<<<lgrid, lblk, 0, stream>>>(S1f, Pjh, g2, be2, S2f, S2h);

  // ---- FFN + add&norm ----
  gemm_f16_kernel<<<ggrid, gblk, 0, stream>>>(S2h, Wt[8], bfb, Pjh, MROWS, D_MODEL, D_MODEL);
  add_ln_kernel<<<lgrid, lblk, 0, stream>>>(S2f, Pjh, g3, be3, S3f, S1h /*scratch*/);
}
